// CriticNetwork_88648124989486
// MI455X (gfx1250) — matrix operand order repaired and checked by reading, not compiled, not run on hardware
//
#include <hip/hip_runtime.h>

typedef __attribute__((ext_vector_type(16))) _Float16 v16h;
typedef __attribute__((ext_vector_type(8)))  float    v8f;

#define BATCH   524288
#define TSTEPS  3
#define DIM     18
#define XROW    (TSTEPS * DIM)   // 54 floats per batch row
#define HID     64
#define GATES   256              // 4*HID
#define WAVES   8
#define RPW     16               // batch rows per wave
#define RPB     (WAVES * RPW)    // 128 rows per block
#define NTILES  (GATES / 16)     // 16 N-tiles of the gate GEMM

// Native CDNA5 v_tanh_f32 (TRANS pipe, 1 instruction, branch-free)
__device__ __forceinline__ float tanh_fast(float x) {
#if __has_builtin(__builtin_amdgcn_tanhf)
  return __builtin_amdgcn_tanhf(x);
#else
  float r;
  asm volatile("v_tanh_f32 %0, %1" : "=v"(r) : "v"(x));
  return r;
#endif
}

// sigmoid(x) = 0.5*tanh(x/2) + 0.5 : one TRANS op instead of exp+rcp
__device__ __forceinline__ float sigmoid_fast(float x) {
  return __builtin_fmaf(0.5f, tanh_fast(0.5f * x), 0.5f);
}

__global__ __launch_bounds__(256, 1)
void critic_lstm_kernel(const float* __restrict__ state_seq,
                        const float* __restrict__ W_ih,
                        const float* __restrict__ W_hh,
                        const float* __restrict__ b_ih,
                        const float* __restrict__ b_hh,
                        const float* __restrict__ W1,
                        const float* __restrict__ b1,
                        const float* __restrict__ W2,
                        const float* __restrict__ b2,
                        float* __restrict__ out)
{
  // Prepacked weight fragments in WMMA B-matrix layout (f16)
  __shared__ __align__(32) _Float16 s_wih[NTILES][32][16];     // 16 KB, K padded 18->32
  __shared__ __align__(32) _Float16 s_whh[NTILES][2][32][16];  // 32 KB, K=64 -> 2 chunks
  __shared__ __align__(32) _Float16 s_w1 [2][2][32][16];       //  4 KB
  __shared__ __align__(16) float    s_x  [RPB * XROW];         // ~27 KB staged input
  // Hidden state kept directly in WMMA A-fragment layout: [wave][k-chunk][lane][elem]
  __shared__ __align__(32) _Float16 s_ha [WAVES][2][32][16];   //  2 KB/wave

  const int tid  = threadIdx.x;
  const int wave = tid >> 5;
  const int lane = tid & 31;
  const int hl   = lane >> 4;   // high half of wave
  const int ll   = lane & 15;

  // ---- prepack weights into LDS B-fragments (52 fragments x 32 lanes) ----
  for (int p = tid; p < 52 * 32; p += 256) {
    const int frag = p >> 5;
    const int vl   = p & 31;
    const int vhl  = vl >> 4;
    const int vll  = vl & 15;
    _Float16* dst;
    if (frag < 16)      dst = &s_wih[frag][vl][0];
    else if (frag < 48) dst = &s_whh[(frag - 16) >> 1][(frag - 16) & 1][vl][0];
    else                dst = &s_w1 [(frag - 48) >> 1][(frag - 48) & 1][vl][0];
#pragma unroll
    for (int e = 0; e < 16; ++e) {
      // B 32x16 layout: lanes 0-15 hold K={0-7,16-23}, lanes 16-31 hold K={8-15,24-31}
      const int kb = (e < 8 ? e : e + 8) + (vhl ? 8 : 0);
      float v = 0.0f;
      if (frag < 16) {                       // B = W_ih^T, tile j = frag
        if (kb < DIM) v = W_ih[(frag * 16 + vll) * DIM + kb];
      } else if (frag < 48) {                // B = W_hh^T, tile (j, ks)
        const int j  = (frag - 16) >> 1;
        const int ks = (frag - 16) & 1;
        v = W_hh[(j * 16 + vll) * HID + ks * 32 + kb];
      } else {                               // B = W1^T, tile (jz, ks)
        const int jz = (frag - 48) >> 1;
        const int ks = (frag - 48) & 1;
        v = W1[(jz * 16 + vll) * HID + ks * 32 + kb];
      }
      dst[e] = (_Float16)v;
    }
  }

  // ---- stage x (vectorized b128, block slab is 16B aligned) ----
  const long long row0 = (long long)blockIdx.x * RPB;
  {
    const float4* gx4 = (const float4*)(state_seq + row0 * XROW);
    float4* sx4 = (float4*)s_x;
    for (int i = tid; i < (RPB * XROW) / 4; i += 256) sx4[i] = gx4[i];
  }

  __syncthreads();

  const float* xrow = &s_x[(wave * RPW + ll) * XROW]; // A-row for this lane

  v8f cacc[4];   // f32 cell state, 4 column tiles x 8 rows per lane

  // One LSTM timestep over the wave's 16x64 tile. first==true exploits h==0,c==0:
  // skips both h-WMMAs per tile and the entire f-gate path (f*c == 0).
  auto lstm_step = [&](const v16h ax, const v16h ah0, const v16h ah1,
                       const bool first) __attribute__((always_inline)) {
#pragma unroll
    for (int jj = 0; jj < 4; ++jj) {          // hidden columns 16*jj .. 16*jj+15
      v8f acc[4];
#pragma unroll
      for (int gt = 0; gt < 4; ++gt) {        // gate order i,f,g,o
        if (first && gt == 1) continue;       // f-gate unused when c==0
        const int j = jj + 4 * gt;
        const float bias = b_ih[j * 16 + ll] + b_hh[j * 16 + ll];
        v8f a = {bias, bias, bias, bias, bias, bias, bias, bias};
        const v16h bx = *(const v16h*)&s_wih[j][lane][0];
        a = __builtin_amdgcn_wmma_f32_16x16x32_f16(false, ax, false, bx, (short)0, a, false, false);
        if (!first) {
          const v16h bh0 = *(const v16h*)&s_whh[j][0][lane][0];
          const v16h bh1 = *(const v16h*)&s_whh[j][1][lane][0];
          a = __builtin_amdgcn_wmma_f32_16x16x32_f16(false, ah0, false, bh0, (short)0, a, false, false);
          a = __builtin_amdgcn_wmma_f32_16x16x32_f16(false, ah1, false, bh1, (short)0, a, false, false);
        }
        acc[gt] = a;
      }
      // LSTM cell update directly on C/D layout: lane->N (col), vgpr->M (row)
      const int c_col = jj * 16 + ll;          // hidden column 0..63
      const int chunk = c_col >> 5;            // K-chunk 0/1
      const int kc    = c_col & 31;
      const int half  = (kc >> 3) & 1;         // which reader lane-half owns K
      const int elem  = (kc & 7) + ((kc & 16) >> 1);
      _Float16* hdst = &s_ha[wave][chunk][half * 16][elem];
#pragma unroll
      for (int v = 0; v < 8; ++v) {
        const float ig = sigmoid_fast(acc[0][v]);
        const float gg = tanh_fast(acc[2][v]);
        const float og = sigmoid_fast(acc[3][v]);
        float c;
        if (first) {
          c = ig * gg;
        } else {
          const float fg = sigmoid_fast(acc[1][v]);
          c = fg * cacc[jj][v] + ig * gg;
        }
        cacc[jj][v] = c;
        const float hv = og * tanh_fast(c);
        hdst[(v + 8 * hl) * 16] = (_Float16)hv;   // row M = v + 8*hl
      }
    }
  };

  // Build x_t A-fragment (16x32, zero padded beyond K=18).
  // A 16x32 layout: lanes 0-15 K={0-7,16-23}; lanes 16-31 K={8-15,24-31}
  auto build_ax = [&](int t) __attribute__((always_inline)) -> v16h {
    v16h ax;
#pragma unroll
    for (int e = 0; e < 16; ++e) {
      const int ka = (e < 8 ? e : e + 8) + (hl ? 8 : 0);
      ax[e] = (ka < DIM) ? (_Float16)xrow[t * DIM + ka] : (_Float16)0.0f;
    }
    return ax;
  };

  // ---- t = 0 peeled: h == 0, c == 0 ----
  {
    const v16h ax0 = build_ax(0);
    lstm_step(ax0, ax0, ax0, true);   // ah operands unused
  }
  __syncthreads();

  // ---- t = 1 .. T-1 ----
  for (int t = 1; t < TSTEPS; ++t) {
    const v16h ax  = build_ax(t);
    const v16h ah0 = *(const v16h*)&s_ha[wave][0][lane][0];
    const v16h ah1 = *(const v16h*)&s_ha[wave][1][lane][0];
    lstm_step(ax, ah0, ah1, false);
    __syncthreads();   // h fragments fully written before next reads
  }

  // ---- MLP head: z = relu(h @ W1^T + b1) via WMMA ----
  const v16h ah0 = *(const v16h*)&s_ha[wave][0][lane][0];
  const v16h ah1 = *(const v16h*)&s_ha[wave][1][lane][0];
  v8f z[2];
#pragma unroll
  for (int jz = 0; jz < 2; ++jz) {
    const float bias = b1[jz * 16 + ll];
    v8f a = {bias, bias, bias, bias, bias, bias, bias, bias};
    const v16h bz0 = *(const v16h*)&s_w1[jz][0][lane][0];
    const v16h bz1 = *(const v16h*)&s_w1[jz][1][lane][0];
    a = __builtin_amdgcn_wmma_f32_16x16x32_f16(false, ah0, false, bz0, (short)0, a, false, false);
    a = __builtin_amdgcn_wmma_f32_16x16x32_f16(false, ah1, false, bz1, (short)0, a, false, false);
#pragma unroll
    for (int v = 0; v < 8; ++v) a[v] = fmaxf(a[v], 0.0f);
    z[jz] = a;
  }

  // ---- value = z @ W2^T + b2 : scale partials, reduce across 16 lanes ----
  const float w2a = W2[ll];
  const float w2b = W2[ll + 16];
  float part[8];
#pragma unroll
  for (int v = 0; v < 8; ++v) part[v] = z[0][v] * w2a + z[1][v] * w2b;
#pragma unroll
  for (int off = 1; off < 16; off <<= 1) {
#pragma unroll
    for (int v = 0; v < 8; ++v) part[v] += __shfl_xor(part[v], off, 32);
  }

  if (ll == 0) {   // lanes 0 and 16 hold rows 0-7 and 8-15 of the wave tile
    const float bb = b2[0];
    float4* o = (float4*)(out + row0 + wave * RPW + hl * 8);  // 32B aligned
    o[0] = make_float4(part[0] + bb, part[1] + bb, part[2] + bb, part[3] + bb);
    o[1] = make_float4(part[4] + bb, part[5] + bb, part[6] + bb, part[7] + bb);
  }
}

extern "C" void kernel_launch(void* const* d_in, const int* in_sizes, int n_in,
                              void* d_out, int out_size, void* d_ws, size_t ws_size,
                              hipStream_t stream) {
  const float* state_seq = (const float*)d_in[0];
  const float* W_ih      = (const float*)d_in[1];
  const float* W_hh      = (const float*)d_in[2];
  const float* b_ih      = (const float*)d_in[3];
  const float* b_hh      = (const float*)d_in[4];
  const float* W1        = (const float*)d_in[5];
  const float* b1        = (const float*)d_in[6];
  const float* W2        = (const float*)d_in[7];
  const float* b2        = (const float*)d_in[8];
  float* out             = (float*)d_out;

  dim3 grid(BATCH / RPB);   // 4096 blocks of 8 waves
  dim3 block(256);
  hipLaunchKernelGGL(critic_lstm_kernel, grid, block, 0, stream,
                     state_seq, W_ih, W_hh, b_ih, b_hh, W1, b1, W2, b2, out);
}
